// Gen_GNN_88656714924073
// MI455X (gfx1250) — matrix operand order repaired and checked by reading, not compiled, not run on hardware
//
#include <hip/hip_runtime.h>

typedef __attribute__((ext_vector_type(16))) __bf16 v16bf;
typedef __attribute__((ext_vector_type(8)))  float  v8f;

#define TM   32
#define TK   32
#define FSTR 56            // bf16 per LDS row slot (112 B: 16B-aligned, 28-DWORD stride -> conflict-free)
#define MP   224           // padded rows (7 x 32 tiles, >= 200)
#define KP   224           // padded K for K=200
#define KH   64            // K for hidden dim

__device__ __forceinline__ __bf16 f2bf(float f) {
    unsigned u = __builtin_bit_cast(unsigned, f);
    u += 0x7fffu + ((u >> 16) & 1u);          // round-to-nearest-even
    unsigned short s = (unsigned short)(u >> 16);
    return __builtin_bit_cast(__bf16, s);
}

__device__ __forceinline__ unsigned pcg(unsigned x) {
    x ^= x >> 16; x *= 0x7feb352du;
    x ^= x >> 15; x *= 0x846ca68bu;
    x ^= x >> 16; return x;
}
__device__ __forceinline__ float u01(unsigned h) { return (float)(h >> 8) * (1.0f / 16777216.0f); }

__device__ __forceinline__ float block_reduce_sum(float v, float* s) {
    int t = threadIdx.x;
    s[t] = v; __syncthreads();
    for (int o = 128; o > 0; o >>= 1) {
        if (t < o) s[t] += s[t + o];
        __syncthreads();
    }
    float r = s[0];
    __syncthreads();
    return r;
}

__device__ __forceinline__ v16bf ld_frag(const __bf16* p) {
    struct P { uint4 lo, hi; } t;                 // 32 B, 16B-aligned -> 2x ds_load_b128
    t.lo = *(const uint4*)p;
    t.hi = *(const uint4*)(p + 8);
    return __builtin_bit_cast(v16bf, t);
}

__device__ __forceinline__ void async_cp16(unsigned lds_addr, const __bf16* src) {
    asm volatile("global_load_async_to_lds_b128 %0, %1, off"
                 :: "v"(lds_addr), "v"(src) : "memory");
}
__device__ __forceinline__ void async_cp8(unsigned lds_addr, const __bf16* src) {
    asm volatile("global_load_async_to_lds_b64 %0, %1, off"
                 :: "v"(lds_addr), "v"(src) : "memory");
}

// ---------------------------------------------------------------------------
// Batched GEMM: C[b](MxN) = A[b](M x Kp, bf16 row-major, zero-padded) *
//                           Bt[b](N x Kp, bf16 K-major, zero-padded)
// SUB = wmma per wave (N block tile = SUB*64). Branch-free double-buffered
// async staging; bias+relu fused; optional f32 out Cf and bf16-transposed Ct.
// ---------------------------------------------------------------------------
template<int SUB>
__global__ __launch_bounds__(256)
void gemm_wmma(const __bf16* __restrict__ A, long long sA,
               const __bf16* __restrict__ Bt, long long sB,
               const float* __restrict__ bias,
               float* __restrict__ Cf, long long sCf,
               __bf16* __restrict__ Ct, long long sCt, int ldCt,
               int M, int N, int Kp, int relu)
{
    constexpr int TNB = SUB * 64;
    __shared__ __bf16 As[2][TM  * FSTR];
    __shared__ __bf16 Bs[2][TNB * FSTR];

    const int b    = blockIdx.z;
    const int m0   = blockIdx.x * TM;
    const int n0   = blockIdx.y * TNB;
    const int tid  = threadIdx.x;
    const int lane = tid & 31;
    const int wave = tid >> 5;
    const int wm   = wave & 1;
    const int wn   = wave >> 1;                   // 0..3
    const int hl   = lane >> 4;
    const int l15  = lane & 15;

    const __bf16* Ab = A  + (long long)b * sA;
    const __bf16* Bb = Bt + (long long)b * sB;

    // A staging: one 8B group per thread (full 32x32 bf16 tile), K in natural order.
    const int ar  = tid >> 3;                     // row 0..31
    const int aq  = tid & 7;                      // quad: K = 4*aq
    const int akg = aq >> 1, ah = aq & 1;
    const int apg = akg;                                      // K-groups in natural order
    const unsigned aoff = (unsigned)(ar * (FSTR * 2) + apg * 16 + ah * 8);
    const unsigned aLds[2] = { (unsigned)(unsigned long long)&As[0][0] + aoff,
                               (unsigned)(unsigned long long)&As[1][0] + aoff };
    const __bf16* aSrc = Ab + (long long)(m0 + ar) * Kp + 4 * aq;

    const unsigned bLds[2] = { (unsigned)(unsigned long long)&Bs[0][0],
                               (unsigned)(unsigned long long)&Bs[1][0] };

    v8f acc[SUB];
    #pragma unroll
    for (int u = 0; u < SUB; ++u)
        #pragma unroll
        for (int i = 0; i < 8; ++i) acc[u][i] = 0.0f;

    const int KT = Kp >> 5;

    // prologue: chunk 0 -> buffer 0  (every wave issues exactly 1+SUB asyncs)
    async_cp8(aLds[0], aSrc);
    #pragma unroll
    for (int i = 0; i < SUB; ++i) {
        const int gid = tid + i * 256;
        const int col = gid >> 2, g = gid & 3;
        async_cp16(bLds[0] + (unsigned)(col * (FSTR * 2) + g * 16),
                   Bb + (long long)(n0 + col) * Kp + 8 * g);
    }

    for (int kt = 0; kt < KT; ++kt) {
        asm volatile("s_wait_asynccnt 0x0" ::: "memory");
        __syncthreads();
        if (kt + 1 < KT) {                        // stage next chunk -> alt buffer
            const int k1 = (kt + 1) * TK;
            const int bi = (kt + 1) & 1;
            async_cp8(aLds[bi], aSrc + k1);
            #pragma unroll
            for (int i = 0; i < SUB; ++i) {
                const int gid = tid + i * 256;
                const int col = gid >> 2, g = gid & 3;
                async_cp16(bLds[bi] + (unsigned)(col * (FSTR * 2) + g * 16),
                           Bb + (long long)(n0 + col) * Kp + k1 + 8 * g);
            }
        }
        const __bf16* AsC = &As[kt & 1][0];
        const __bf16* BsC = &Bs[kt & 1][0];
        const v16bf af = ld_frag(AsC + (wm * 16 + l15) * FSTR + hl * 16);
        #pragma unroll
        for (int u = 0; u < SUB; ++u) {
            const v16bf bf = ld_frag(BsC + (wn * (SUB * 16) + u * 16 + l15) * FSTR + hl * 16);
            acc[u] = __builtin_amdgcn_wmma_f32_16x16x32_bf16(false, af, false, bf,
                                                             (short)0, acc[u], false, false);
        }
    }

    // epilogue: C/D layout (VGPR r -> M = r + hl*8)
    #pragma unroll
    for (int u = 0; u < SUB; ++u) {
        const int col = n0 + wn * (SUB * 16) + u * 16 + l15;
        if (col < N) {
            const float bv = bias ? bias[col] : 0.0f;
            #pragma unroll
            for (int r = 0; r < 8; ++r) {
                const int row = m0 + wm * 16 + r + hl * 8;
                if (row < M) {
                    float v = acc[u][r] + bv;
                    if (relu) v = fmaxf(v, 0.0f);
                    if (Cf) Cf[(long long)b * sCf + (long long)row * N + col] = v;
                    if (Ct) Ct[(long long)b * sCt + (long long)col * ldCt + row] = f2bf(v);
                }
            }
        }
    }
}

// ---------------------------------------------------------------------------
__global__ __launch_bounds__(256)
void row_dis(const float* __restrict__ adj, float* __restrict__ dis, int N)
{
    __shared__ float sbuf[256];
    const int row = blockIdx.x;
    const int b = row / N, i = row % N;
    float sum = 0.0f;
    for (int j = threadIdx.x; j < N; j += 256)
        if (j != i) sum += adj[((long long)b * N + i) * N + j];
    float tot = block_reduce_sum(sum, sbuf) + 1.0f;
    if (threadIdx.x == 0) dis[row] = rsqrtf(fmaxf(tot, 1.0f));
}

__global__ __launch_bounds__(256)
void adj_norm_bf(const float* __restrict__ adj, const float* __restrict__ dis,
                 __bf16* __restrict__ out, int N, long long total)
{
    for (long long idx = (long long)blockIdx.x * 256 + threadIdx.x;
         idx < total; idx += (long long)gridDim.x * 256) {
        long long bb = idx / ((long long)N * N);
        int r = (int)((idx / N) % N);
        int cc = (int)(idx % N);
        float a = (r == cc) ? 1.0f : adj[idx];
        out[(bb * MP + r) * KP + cc] = f2bf(dis[bb * N + r] * a * dis[bb * N + cc]);
    }
}

__global__ __launch_bounds__(256)
void adj_sample(const float* __restrict__ xh, const float* __restrict__ adj,
                float* __restrict__ adj_new, float* __restrict__ loss,
                long long total, unsigned seed)
{
    __shared__ float sbuf[256];
    float part = 0.0f;
    for (long long idx = (long long)blockIdx.x * 256 + threadIdx.x;
         idx < total; idx += (long long)gridDim.x * 256) {
        float p = 1.0f / (1.0f + __expf(-xh[idx]));
        unsigned hb = (unsigned)idx * 0x9E3779B9u + seed;
        float m4 = 0.25f * (u01(pcg(hb)) + u01(pcg(hb + 1u)) +
                            u01(pcg(hb + 2u)) + u01(pcg(hb + 3u)));
        float V  = 0.5f + (m4 - 0.5f) * 0.2f;
        float xs = 1.0f / (1.0f + __expf(-(V + p - 1.0f) * 10.0f));
        adj_new[idx] = adj[idx] * xs;
        float d = p - 0.5f;
        part += 0.5f * d * d;
    }
    float tot = block_reduce_sum(part, sbuf);
    if (threadIdx.x == 0) atomicAdd(loss, tot);
}

__global__ __launch_bounds__(256)
void bn_nodes(const float* __restrict__ X, const float* __restrict__ g,
              const float* __restrict__ be, float* __restrict__ Yf,
              __bf16* __restrict__ Yb, int ldYb,
              int Bc, int N, int C, int relu_out)
{
    __shared__ float sbuf[256];
    const int n = blockIdx.x;
    const int total = Bc * C;
    float sum = 0.0f, sq = 0.0f;
    for (int e = threadIdx.x; e < total; e += 256) {
        int b = e / C, cc = e % C;
        float v = X[((long long)b * N + n) * C + cc];
        sum += v; sq += v * v;
    }
    float S  = block_reduce_sum(sum, sbuf);
    float S2 = block_reduce_sum(sq, sbuf);
    float m   = S / (float)total;
    float var = S2 / (float)total - m * m;
    float sc  = g[n] * rsqrtf(var + 1e-5f);
    float sh  = be[n] - m * sc;
    for (int e = threadIdx.x; e < total; e += 256) {
        int b = e / C, cc = e % C;
        long long off = ((long long)b * N + n) * C + cc;
        float v = X[off] * sc + sh;
        if (relu_out) v = fmaxf(v, 0.0f);
        if (Yf) Yf[off] = v;
        if (Yb) Yb[((long long)b * MP + n) * ldYb + cc] = f2bf(v);
    }
}

__global__ __launch_bounds__(256)
void cvt_pad(const float* __restrict__ in, __bf16* __restrict__ out,
             int N, long long total)
{
    for (long long idx = (long long)blockIdx.x * 256 + threadIdx.x;
         idx < total; idx += (long long)gridDim.x * 256) {
        long long bb = idx / ((long long)N * N);
        int r = (int)((idx / N) % N);
        int cc = (int)(idx % N);
        out[(bb * MP + r) * KP + cc] = f2bf(in[idx]);
    }
}

__global__ __launch_bounds__(256)
void cvt_bt(const float* __restrict__ in, __bf16* __restrict__ out,
            int K, int N, int ldW)
{
    long long total = (long long)K * N;
    for (long long idx = (long long)blockIdx.x * 256 + threadIdx.x;
         idx < total; idx += (long long)gridDim.x * 256) {
        int k = (int)(idx / N), n = (int)(idx % N);
        out[(long long)n * ldW + k] = f2bf(in[idx]);
    }
}

__global__ __launch_bounds__(256)
void head_kernel(const float* __restrict__ X, const float* __restrict__ Wl1,
                 const float* __restrict__ bl1, const float* __restrict__ Wl2,
                 const float* __restrict__ bl2, float* __restrict__ out,
                 int N, int C)
{
    __shared__ float sbuf[256];
    const int b = blockIdx.x;
    const int n = threadIdx.x;
    float contrib = 0.0f;
    if (n < N) {
        float acc = 0.0f;
        for (int cc = 0; cc < C; ++cc) {
            long long off = ((long long)b * N + n) * C + cc;
            float v = X[off];
            unsigned h = pcg((unsigned)off ^ 0xDEADBEEFu);
            v = (h & 1u) ? v * 2.0f : 0.0f;
            acc += v * Wl1[cc];
        }
        acc = fmaxf(acc + bl1[0], 0.0f);
        contrib = acc * Wl2[n];
    }
    float s = block_reduce_sum(contrib, sbuf);
    if (threadIdx.x == 0) out[b] = fmaxf(s + bl2[0], 0.0f);
}

__global__ void zero_f(float* p) { *p = 0.0f; }
__global__ void copy_f(const float* s, float* d) { *d = *s; }

// ---------------------------------------------------------------------------
extern "C" void kernel_launch(void* const* d_in, const int* in_sizes, int n_in,
                              void* d_out, int out_size, void* d_ws, size_t ws_size,
                              hipStream_t stream)
{
    (void)in_sizes; (void)n_in; (void)out_size; (void)ws_size;
    const int B = 32, N = 200, H = 64, F2 = 128;
    const long long BNN = (long long)B * N * N;

    const float* x_in   = (const float*)d_in[0];
    const float* adj_in = (const float*)d_in[1];
    const float* Wa[2]  = { (const float*)d_in[2],  (const float*)d_in[4]  };
    const float* ba[2]  = { (const float*)d_in[3],  (const float*)d_in[5]  };
    const float* W1s[2] = { (const float*)d_in[6],  (const float*)d_in[14] };
    const float* b1s[2] = { (const float*)d_in[7],  (const float*)d_in[15] };
    const float* g1s[2] = { (const float*)d_in[8],  (const float*)d_in[16] };
    const float* be1s[2]= { (const float*)d_in[9],  (const float*)d_in[17] };
    const float* W2s[2] = { (const float*)d_in[10], (const float*)d_in[18] };
    const float* b2s[2] = { (const float*)d_in[11], (const float*)d_in[19] };
    const float* g2s[2] = { (const float*)d_in[12], (const float*)d_in[20] };
    const float* be2s[2]= { (const float*)d_in[13], (const float*)d_in[21] };
    const float* Wl1 = (const float*)d_in[22];
    const float* bl1 = (const float*)d_in[23];
    const float* Wl2 = (const float*)d_in[24];
    const float* bl2 = (const float*)d_in[25];

    float* out = (float*)d_out;
    char*  w   = (char*)d_ws;
    size_t o = 0;
    auto allocf = [&](long long n) { float*  p = (float*) (w + o); o += (size_t)n * 4; o = (o + 63) & ~63ull; return p; };
    auto allocb = [&](long long n) { __bf16* p = (__bf16*)(w + o); o += (size_t)n * 2; o = (o + 63) & ~63ull; return p; };

    float*  t1   = allocf(BNN);
    float*  adjA = allocf(BNN);
    float*  adjB = allocf(BNN);
    float*  xbuf = allocf(BNN);
    float*  dis  = allocf((long long)B * N);
    float*  loss = allocf(16);

    const size_t bfStart = o;
    const long long sAfull = (long long)MP * KP;   // 50176
    const long long sH     = (long long)MP * KH;   // 14336
    const long long sT0T   = 256LL * KP;           // 57344
    __bf16* adjn_b = allocb((long long)B * sAfull);
    __bf16* x_b    = allocb((long long)B * sAfull);
    __bf16* xb_b   = allocb((long long)B * sAfull);
    __bf16* h1_b   = allocb((long long)B * sH);
    __bf16* t0T    = allocb((long long)B * sT0T);
    __bf16* WaT[2] = { allocb(256LL * KP), allocb(256LL * KP) };
    __bf16* W1T[2] = { allocb(64LL * KP),  allocb(64LL * KP)  };
    __bf16* W2T[2] = { allocb(256LL * KH), allocb(128LL * KH) };
    const size_t bfBytes = o - bfStart;

    hipMemsetAsync(w + bfStart, 0, bfBytes, stream);
    zero_f<<<1, 1, 0, stream>>>(loss);

    cvt_bt<<<160, 256, 0, stream>>>(Wa[0], WaT[0], N, N, KP);
    cvt_bt<<<160, 256, 0, stream>>>(Wa[1], WaT[1], N, N, KP);
    cvt_bt<<<64, 256, 0, stream>>>(W1s[0], W1T[0], N, H, KP);
    cvt_bt<<<64, 256, 0, stream>>>(W1s[1], W1T[1], N, H, KP);
    cvt_bt<<<64, 256, 0, stream>>>(W2s[0], W2T[0], H, N,  KH);
    cvt_bt<<<64, 256, 0, stream>>>(W2s[1], W2T[1], H, F2, KH);
    cvt_pad<<<2048, 256, 0, stream>>>(x_in, x_b, N, BNN);

    const int elemBlocks = (int)((BNN + 255) / 256);
    const float*  adjcur = adj_in;
    const __bf16* xA = x_b;
    float* adjbuf[2] = { adjA, adjB };
    const dim3 g7(7, 1, B);

    for (int s = 0; s < 2; ++s) {
        row_dis    <<<B * N, 256, 0, stream>>>(adjcur, dis, N);
        adj_norm_bf<<<elemBlocks, 256, 0, stream>>>(adjcur, dis, adjn_b, N, BNN);
        gemm_wmma<4><<<g7, 256, 0, stream>>>(xA, sAfull, WaT[s], 0, nullptr,
                                             nullptr, 0, t0T, sT0T, KP, N, N, KP, 0);
        gemm_wmma<4><<<g7, 256, 0, stream>>>(adjn_b, sAfull, t0T, sT0T, ba[s],
                                             t1, (long long)N * N, nullptr, 0, 0, N, N, KP, 0);
        adj_sample<<<elemBlocks, 256, 0, stream>>>(t1, adjcur, adjbuf[s], loss,
                                                   BNN, 0x12345u + 0x9E3779B9u * (unsigned)s);
        adjcur = adjbuf[s];

        row_dis    <<<B * N, 256, 0, stream>>>(adjcur, dis, N);
        adj_norm_bf<<<elemBlocks, 256, 0, stream>>>(adjcur, dis, adjn_b, N, BNN);

        gemm_wmma<1><<<g7, 256, 0, stream>>>(xA, sAfull, W1T[s], 0, nullptr,
                                             nullptr, 0, t0T, sT0T, KP, N, H, KP, 0);
        gemm_wmma<1><<<g7, 256, 0, stream>>>(adjn_b, sAfull, t0T, sT0T, b1s[s],
                                             t1, (long long)N * H, nullptr, 0, 0, N, H, KP, 1);
        bn_nodes<<<N, 256, 0, stream>>>(t1, g1s[s], be1s[s], nullptr, h1_b, KH, B, N, H, 0);

        const int Co = (s == 0) ? N : F2;
        if (s == 0) {
            gemm_wmma<4><<<g7, 256, 0, stream>>>(h1_b, sH, W2T[s], 0, nullptr,
                                                 nullptr, 0, t0T, sT0T, KP, N, Co, KH, 0);
            gemm_wmma<4><<<g7, 256, 0, stream>>>(adjn_b, sAfull, t0T, sT0T, b2s[s],
                                                 t1, (long long)N * Co, nullptr, 0, 0, N, Co, KP, 1);
        } else {
            gemm_wmma<2><<<g7, 256, 0, stream>>>(h1_b, sH, W2T[s], 0, nullptr,
                                                 nullptr, 0, t0T, sT0T, KP, N, Co, KH, 0);
            gemm_wmma<2><<<g7, 256, 0, stream>>>(adjn_b, sAfull, t0T, sT0T, b2s[s],
                                                 t1, (long long)N * Co, nullptr, 0, 0, N, Co, KP, 1);
        }
        bn_nodes<<<N, 256, 0, stream>>>(t1, g2s[s], be2s[s], xbuf, xb_b, KP, B, N, Co, 1);
        xA = xb_b;
    }

    head_kernel<<<B, 256, 0, stream>>>(xbuf, Wl1, bl1, Wl2, bl2, out, N, F2);
    copy_f<<<1, 1, 0, stream>>>(loss, out + 32);
}
